// Classifier_22299470201420
// MI455X (gfx1250) — 16-bit and 8-bit matrix operand order repaired and checked by reading, compiled, not run on hardware
//
#include <hip/hip_runtime.h>

typedef __attribute__((ext_vector_type(16))) _Float16 v16h;
typedef __attribute__((ext_vector_type(8)))  _Float16 v8h;
typedef __attribute__((ext_vector_type(4)))  _Float16 v4h;
typedef __attribute__((ext_vector_type(8)))  float    v8f;
typedef __attribute__((ext_vector_type(8)))  int      v8i;

constexpr int B_  = 4096;
constexpr int F_  = 1024;
constexpr int D_  = 10000;
constexpr int D2_ = 10048;
constexpr int C_  = 100;
constexpr int C2_ = 112;

__global__ void cvt_S(const float* __restrict__ src, _Float16* __restrict__ dst) {
    int i = blockIdx.x * blockDim.x + threadIdx.x;
    if (i >= B_ * F_ / 4) return;
    float4 f = reinterpret_cast<const float4*>(src)[i];
    v4h h;
    h[0] = (_Float16)(f.x - 0.5f);
    h[1] = (_Float16)(f.y - 0.5f);
    h[2] = (_Float16)(f.z - 0.5f);
    h[3] = (_Float16)(f.w - 0.5f);
    reinterpret_cast<v4h*>(dst)[i] = h;
}

__global__ void cvt_W(const float* __restrict__ src, _Float16* __restrict__ dst) {
    int i = blockIdx.x * blockDim.x + threadIdx.x;
    if (i >= D_ * F_ / 4) return;
    float4 f = reinterpret_cast<const float4*>(src)[i];
    v4h h;
    h[0] = (_Float16)f.x;
    h[1] = (_Float16)f.y;
    h[2] = (_Float16)f.z;
    h[3] = (_Float16)f.w;
    reinterpret_cast<v4h*>(dst)[i] = h;
}

__global__ void build_cb(const float* __restrict__ cent, char* __restrict__ cb) {
    int i = blockIdx.x * blockDim.x + threadIdx.x;
    if (i >= C2_ * D2_) return;
    int c = i / D2_;
    int d = i - c * D2_;
    char v = 0;
    if (c < C_ && d < D_) v = (cent[(size_t)c * D_ + d] > 0.5f) ? (char)1 : (char)-1;
    cb[i] = v;
}

__global__ void pad_bip(char* __restrict__ bip) {
    int i = blockIdx.x * blockDim.x + threadIdx.x;
    constexpr int PAD = D2_ - D_;
    int row = i / PAD;
    int col = D_ + (i - row * PAD);
    if (row < B_) bip[(size_t)row * D2_ + col] = 0;
}

__global__ void __launch_bounds__(256)
encode_gemm(const _Float16* __restrict__ Sh, const _Float16* __restrict__ Wh,
            char* __restrict__ bip) {
    const int lane = threadIdx.x & 31;
    const int m    = lane & 15;
    const int half = lane >> 4;
    const int h8   = half << 3;
    const int h16  = half << 4;
    const int mBase = blockIdx.x * 512 + (threadIdx.x >> 5) * 64;
    const int nBase = blockIdx.y * 80;

    const _Float16* arow[4];
    #pragma unroll
    for (int i = 0; i < 4; ++i)
        arow[i] = Sh + (size_t)(mBase + i * 16 + m) * F_;
    const _Float16* wrow[5];
    #pragma unroll
    for (int t = 0; t < 5; ++t)
        wrow[t] = Wh + (size_t)(nBase + t * 16 + m) * F_;

    v8f acc[4][5] = {};

    for (int kb = 0; kb < F_; kb += 32) {
        v16h a[4];
        #pragma unroll
        for (int i = 0; i < 4; ++i) {
            v8h lo = *reinterpret_cast<const v8h*>(arow[i] + kb + h16);
            v8h hi = *reinterpret_cast<const v8h*>(arow[i] + kb + h16 + 8);
            a[i] = __builtin_shufflevector(lo, hi,
                     0,1,2,3,4,5,6,7,8,9,10,11,12,13,14,15);
        }
        #pragma unroll
        for (int t = 0; t < 5; ++t) {
            v16h b = *reinterpret_cast<const v16h*>(wrow[t] + kb + h16);
            #pragma unroll
            for (int i = 0; i < 4; ++i)
                acc[i][t] = __builtin_amdgcn_wmma_f32_16x16x32_f16(
                    false, a[i], false, b, (short)0, acc[i][t], false, false);
        }
    }
    #pragma unroll
    for (int i = 0; i < 4; ++i)
        #pragma unroll
        for (int t = 0; t < 5; ++t)
            #pragma unroll
            for (int r = 0; r < 8; ++r) {
                int row = mBase + i * 16 + h8 + r;
                int col = nBase + t * 16 + m;
                bip[(size_t)row * D2_ + col] = (acc[i][t][r] > 0.0f) ? (char)1 : (char)-1;
            }
}

__global__ void __launch_bounds__(256)
hamming_gemm(const char* __restrict__ bip, const char* __restrict__ cb,
             int* __restrict__ out) {
    const int lane = threadIdx.x & 31;
    const int n    = lane & 15;
    const int half = lane >> 4;
    const int mBase = (blockIdx.x * 8 + (threadIdx.x >> 5)) * 16;
    const int nBase = blockIdx.y * 16;

    v8i acc = {};
    const char* arow = bip + (size_t)(mBase + n) * D2_;
    const char* brow = cb  + (size_t)(nBase + n) * D2_;
    for (int kb = 0; kb < D2_; kb += 64) {
        int2 a0 = *reinterpret_cast<const int2*>(arow + kb +      (half << 4));
        int2 a1 = *reinterpret_cast<const int2*>(arow + kb + 8  + (half << 4));
        int2 a2 = *reinterpret_cast<const int2*>(arow + kb + 32 + (half << 4));
        int2 a3 = *reinterpret_cast<const int2*>(arow + kb + 40 + (half << 4));
        int4 b0 = *reinterpret_cast<const int4*>(brow + kb +      (half << 4));
        int4 b1 = *reinterpret_cast<const int4*>(brow + kb + 32 + (half << 4));
        v8i a = {a0.x, a0.y, a1.x, a1.y, a2.x, a2.y, a3.x, a3.y};
        v8i b = {b0.x, b0.y, b0.z, b0.w, b1.x, b1.y, b1.z, b1.w};
        acc = __builtin_amdgcn_wmma_i32_16x16x64_iu8(true, a, true, b, acc,
                                                     false, false);
    }
    int col = nBase + n;
    if (col < C_) {
        #pragma unroll
        for (int r = 0; r < 8; ++r) {
            int row = mBase + (half << 3) + r;
            out[(size_t)row * C_ + col] = (D_ + acc[r]) >> 1;
        }
    }
}

extern "C" void kernel_launch(void* const* d_in, const int* in_sizes, int n_in,
                              void* d_out, int out_size, void* d_ws, size_t ws_size,
                              hipStream_t stream) {
    const float* samples   = (const float*)d_in[0];
    const float* W         = (const float*)d_in[1];
    const float* centroids = (const float*)d_in[2];
    int* out = (int*)d_out;

    char*     bip = (char*)d_ws;
    char*     cb  = bip + (size_t)B_ * D2_;
    _Float16* Sh  = (_Float16*)(cb + (size_t)C2_ * D2_);
    _Float16* Wh  = Sh + (size_t)B_ * F_;

    cvt_S   <<<(B_ * F_ / 4 + 255) / 256, 256, 0, stream>>>(samples, Sh);
    cvt_W   <<<(D_ * F_ / 4 + 255) / 256, 256, 0, stream>>>(W, Wh);
    build_cb<<<(C2_ * D2_ + 255) / 256, 256, 0, stream>>>(centroids, cb);
    pad_bip <<<(B_ * (D2_ - D_) + 255) / 256, 256, 0, stream>>>(bip);
    encode_gemm <<<dim3(B_ / 512, D_ / 80), 256, 0, stream>>>(Sh, Wh, bip);
    hamming_gemm<<<dim3(B_ / 128, C2_ / 16), 256, 0, stream>>>(bip, cb, out);
}
